// MultiHeadAttention_8005819039801
// MI455X (gfx1250) — matrix operand order repaired and checked by reading, not compiled, not run on hardware
//
#include <hip/hip_runtime.h>

// Problem constants (from reference): B=2, S=2048, E=1024, H=16, D=64
#define BB    2
#define S_LEN 2048
#define EMB   1024
#define NH    16
#define DH    64
#define MROWS (BB * S_LEN)   // 4096 token rows

typedef __attribute__((ext_vector_type(16))) __bf16 v16bf;
typedef __attribute__((ext_vector_type(8)))  float  v8f;

union BF16Frag { v16bf v; unsigned int u[8]; };

__device__ __forceinline__ unsigned short f2bf(float f) {
  unsigned int u = __float_as_uint(f);
  u += 0x7FFFu + ((u >> 16) & 1u);     // round-to-nearest-even
  return (unsigned short)(u >> 16);
}

// ---------------------------------------------------------------- converts --
__global__ void k_conv_plain(const float* __restrict__ in,
                             unsigned short* __restrict__ out, int n) {
  int i = blockIdx.x * blockDim.x + threadIdx.x;
  if (i < n) out[i] = f2bf(in[i]);
}

// W[h][e][d] (H,E,D) -> out[e*1024 + h*64 + d]  (row-major [E, H*D])
__global__ void k_conv_wqkv(const float* __restrict__ W,
                            unsigned short* __restrict__ out) {
  int i = blockIdx.x * blockDim.x + threadIdx.x;  // over E*H*D == 1M
  int e   = i >> 10;
  int rem = i & 1023;
  int h   = rem >> 6;
  int d   = rem & 63;
  out[i] = f2bf(W[((size_t)h * EMB + e) * DH + d]);
}

// -------------------------------------------------------------------- GEMM --
// C[M,N] = A[M,K](bf16) * B[K,N](bf16) + bias[N];  M,N,K multiples of 128/32.
// WG = 256 threads = 8 waves. WG tile 128x128; wave tile 64x32 (4x2 WMMA).
template <int STORE_F32>
__global__ __launch_bounds__(256) void k_gemm(
    const unsigned short* __restrict__ A,
    const unsigned short* __restrict__ Bm,
    const float* __restrict__ bias,
    void* __restrict__ Cv,
    int M, int N, int K) {
  __shared__ unsigned short sA[128][32];   // 8 KB
  __shared__ unsigned short sB[32][128];   // 8 KB

  const int tid  = threadIdx.x;
  const int wave = tid >> 5;
  const int lane = tid & 31;
  const int half = lane >> 4;
  const int l16  = lane & 15;
  const int m_base = blockIdx.x * 128;
  const int n_base = blockIdx.y * 128;
  const int wm = (wave >> 2) * 64;         // wave row offset in WG tile
  const int wn = (wave & 3) * 32;          // wave col offset in WG tile

  const v8f zf = {0.f, 0.f, 0.f, 0.f, 0.f, 0.f, 0.f, 0.f};
  v8f acc[4][2];
#pragma unroll
  for (int mi = 0; mi < 4; ++mi)
#pragma unroll
    for (int ni = 0; ni < 2; ++ni) acc[mi][ni] = zf;

  // cooperative staging indices
  const int ar = tid >> 1, ac = (tid & 1) * 16;   // A: 128 rows x 32 cols
  const int br = tid >> 3, bc = (tid & 7) * 16;   // B: 32 rows x 128 cols
  const unsigned short* aptr = A  + (size_t)(m_base + ar) * K + ac;
  const unsigned short* bptr = Bm + (size_t)br * N + n_base + bc;

  for (int k0 = 0; k0 < K; k0 += 32) {
    uint4 a0 = *(const uint4*)(aptr + k0);
    uint4 a1 = *(const uint4*)(aptr + k0 + 8);
    uint4 b0 = *(const uint4*)(bptr + (size_t)k0 * N);
    uint4 b1 = *(const uint4*)(bptr + (size_t)k0 * N + 8);
    if (k0 + 32 < K) {                     // gfx1250 global_prefetch_b8
      __builtin_prefetch(aptr + k0 + 32, 0, 1);
      __builtin_prefetch(bptr + (size_t)(k0 + 32) * N, 0, 1);
    }
    __syncthreads();                       // previous-iter LDS reads done
    *(uint4*)&sA[ar][ac]     = a0;
    *(uint4*)&sA[ar][ac + 8] = a1;
    *(uint4*)&sB[br][bc]     = b0;
    *(uint4*)&sB[br][bc + 8] = b1;
    __syncthreads();

    // A fragments: ISA 16-bit A 16x32 layout (K-pairs, 8*half stagger)
    BF16Frag af[4];
#pragma unroll
    for (int mi = 0; mi < 4; ++mi) {
      const int row = wm + mi * 16 + l16;
#pragma unroll
      for (int vi = 0; vi < 8; ++vi) {
        const int kk = ((vi < 4) ? (vi * 2) : (16 + (vi - 4) * 2)) + 8 * half;
        af[mi].u[vi] = *(const unsigned int*)&sA[row][kk];
      }
    }
    // B fragments: column n = l16 per lane, K in same order as A
    BF16Frag bf[2];
#pragma unroll
    for (int ni = 0; ni < 2; ++ni) {
#pragma unroll
      for (int vi = 0; vi < 8; ++vi) {
        const int kk = ((vi < 4) ? (vi * 2) : (16 + (vi - 4) * 2)) + 8 * half;
        bf[ni].u[vi] = (unsigned int)sB[kk][wn + ni * 16 + l16] |
                       ((unsigned int)sB[kk + 1][wn + ni * 16 + l16] << 16);
      }
    }
#pragma unroll
    for (int mi = 0; mi < 4; ++mi)
#pragma unroll
      for (int ni = 0; ni < 2; ++ni)
        acc[mi][ni] = __builtin_amdgcn_wmma_f32_16x16x32_bf16(
            false, af[mi].v, false, bf[ni].v, (short)0, acc[mi][ni], false,
            false);
  }

  // epilogue: C fragment (row = r + 8*half striped over VGPRs, col = l16)
#pragma unroll
  for (int ni = 0; ni < 2; ++ni) {
    const int col = n_base + wn + ni * 16 + l16;
    const float bb = bias[col];
#pragma unroll
    for (int mi = 0; mi < 4; ++mi) {
#pragma unroll
      for (int r = 0; r < 8; ++r) {
        const int row = m_base + wm + mi * 16 + r + 8 * half;
        const float v = acc[mi][ni][r] + bb;
        if (STORE_F32)
          ((float*)Cv)[(size_t)row * N + col] = v;
        else
          ((unsigned short*)Cv)[(size_t)row * N + col] = f2bf(v);
      }
    }
  }
}

// --------------------------------------------------------- flash attention --
// Grid: B*H*(S/128) WGs; 8 waves/WG; wave owns 16 query rows. KV steps of 32.
__global__ __launch_bounds__(256) void k_flash(
    const unsigned short* __restrict__ Qb,   // [B*S, H*D] bf16
    const unsigned short* __restrict__ Kb,
    const unsigned short* __restrict__ Vb,
    unsigned short* __restrict__ Ob) {       // [B*S, H*D] bf16
  __shared__ unsigned short sKt[DH][32];     // K tile transposed [d][t], 4 KB
  __shared__ unsigned short sP[8][16][32];   // per-wave P staging, 8 KB

  const int tid  = threadIdx.x;
  const int wave = tid >> 5;
  const int lane = tid & 31;
  const int half = lane >> 4;
  const int l16  = lane & 15;

  const int chunks = S_LEN / 128;            // 16
  const int bh = blockIdx.x / chunks;
  const int qc = blockIdx.x % chunks;
  const int b  = bh / NH;
  const int h  = bh % NH;
  const int q0 = qc * 128 + wave * 16;       // first query row of this wave

  // preload Q as two 16x32 A-fragments (d 0..31, d 32..63)
  const size_t qrowoff = (size_t)(b * S_LEN + q0 + l16) * EMB + h * DH;
  BF16Frag qf[2];
#pragma unroll
  for (int kh = 0; kh < 2; ++kh)
#pragma unroll
    for (int vi = 0; vi < 8; ++vi) {
      const int kk =
          ((vi < 4) ? (vi * 2) : (16 + (vi - 4) * 2)) + 8 * half + 32 * kh;
      qf[kh].u[vi] = *(const unsigned int*)(Qb + qrowoff + kk);
    }

  const v8f zf = {0.f, 0.f, 0.f, 0.f, 0.f, 0.f, 0.f, 0.f};
  v8f o[4] = {zf, zf, zf, zf};               // O accumulator 16x64
  float m[8], l[8];
#pragma unroll
  for (int r = 0; r < 8; ++r) { m[r] = -3.0e38f; l[r] = 0.f; }

  const int tEnd = qc * 128 + 127;           // WG-uniform causal bound

  for (int t0 = 0; t0 <= tEnd; t0 += 32) {
    // cooperative staging of K tile, transposed to [d][t]
    {
      const int trow = tid >> 3;             // 0..31
      const int dseg = (tid & 7) * 8;        // 0..56
      const unsigned short* src =
          Kb + (size_t)(b * S_LEN + t0 + trow) * EMB + h * DH + dseg;
      unsigned short tmp[8];
      *(uint4*)tmp = *(const uint4*)src;
#pragma unroll
      for (int j = 0; j < 8; ++j) sKt[dseg + j][trow] = tmp[j];
    }
    __syncthreads();

    if (t0 <= q0 + 15) {                     // wave-uniform causal skip
      // scores S = Q K^T : two 16x16 tiles over t, K-dim = 64 (2 WMMA each)
      v8f s[2];
#pragma unroll
      for (int nt = 0; nt < 2; ++nt) {
        BF16Frag kb0, kb1;
#pragma unroll
        for (int vi = 0; vi < 8; ++vi) {
          const int kk = ((vi < 4) ? (vi * 2) : (16 + (vi - 4) * 2)) + 8 * half;
          kb0.u[vi] = (unsigned int)sKt[kk][nt * 16 + l16] |
                      ((unsigned int)sKt[kk + 1][nt * 16 + l16] << 16);
          kb1.u[vi] = (unsigned int)sKt[32 + kk][nt * 16 + l16] |
                      ((unsigned int)sKt[32 + kk + 1][nt * 16 + l16] << 16);
        }
        v8f c = __builtin_amdgcn_wmma_f32_16x16x32_bf16(
            false, qf[0].v, false, kb0.v, (short)0, zf, false, false);
        c = __builtin_amdgcn_wmma_f32_16x16x32_bf16(
            false, qf[1].v, false, kb1.v, (short)0, c, false, false);
        s[nt] = c;
      }
      // scale + causal mask
#pragma unroll
      for (int nt = 0; nt < 2; ++nt)
#pragma unroll
        for (int r = 0; r < 8; ++r) {
          const int tcol = t0 + nt * 16 + l16;
          const int qrow = q0 + r + 8 * half;
          const float v = s[nt][r] * 0.125f;  // 1/sqrt(64)
          s[nt][r] = (tcol > qrow) ? -3.0e38f : v;
        }
      // online softmax; per-row stats reduced across the 16-lane half
#pragma unroll
      for (int r = 0; r < 8; ++r) {
        float mx = fmaxf(s[0][r], s[1][r]);
#pragma unroll
        for (int off = 1; off < 16; off <<= 1)
          mx = fmaxf(mx, __shfl_xor(mx, off, 32));
        const float mn    = fmaxf(m[r], mx);
        const float alpha = __expf(m[r] - mn);
        const float p0    = __expf(s[0][r] - mn);
        const float p1    = __expf(s[1][r] - mn);
        float rs = p0 + p1;
#pragma unroll
        for (int off = 1; off < 16; off <<= 1) rs += __shfl_xor(rs, off, 32);
        m[r] = mn;
        l[r] = l[r] * alpha + rs;
#pragma unroll
        for (int dn = 0; dn < 4; ++dn) o[dn][r] *= alpha;
        const int prow = r + 8 * half;
        sP[wave][prow][l16]      = f2bf(p0);
        sP[wave][prow][16 + l16] = f2bf(p1);
      }
      // P as 16x32 A-fragment (wave-private LDS, DS ops in order per wave)
      BF16Frag pf;
#pragma unroll
      for (int vi = 0; vi < 8; ++vi) {
        const int kk = ((vi < 4) ? (vi * 2) : (16 + (vi - 4) * 2)) + 8 * half;
        pf.u[vi] = *(const unsigned int*)&sP[wave][l16][kk];
      }
      // O += P V : V B-fragments direct from global (column d per lane, t in A order)
      const unsigned short* vrow =
          Vb + (size_t)(b * S_LEN + t0) * EMB + h * DH;
#pragma unroll
      for (int dn = 0; dn < 4; ++dn) {
        BF16Frag vf;
#pragma unroll
        for (int vi = 0; vi < 8; ++vi) {
          const int kk = ((vi < 4) ? (vi * 2) : (16 + (vi - 4) * 2)) + 8 * half;
          vf.u[vi] = (unsigned int)vrow[(size_t)kk * EMB + dn * 16 + l16] |
                     ((unsigned int)vrow[(size_t)(kk + 1) * EMB + dn * 16 + l16] << 16);
        }
        o[dn] = __builtin_amdgcn_wmma_f32_16x16x32_bf16(
            false, pf.v, false, vf.v, (short)0, o[dn], false, false);
      }
    }
    __syncthreads();
  }

  // normalize and store O (bf16) as [B*S, H*D] (= concat-heads layout)
#pragma unroll
  for (int r = 0; r < 8; ++r) {
    const float inv  = 1.0f / l[r];
    const int   qrow = q0 + r + 8 * half;
#pragma unroll
    for (int dn = 0; dn < 4; ++dn)
      Ob[(size_t)(b * S_LEN + qrow) * EMB + h * DH + dn * 16 + l16] =
          f2bf(o[dn][r] * inv);
  }
}

// ------------------------------------------------------------------ launch --
extern "C" void kernel_launch(void* const* d_in, const int* in_sizes, int n_in,
                              void* d_out, int out_size, void* d_ws,
                              size_t ws_size, hipStream_t stream) {
  (void)in_sizes; (void)n_in; (void)out_size; (void)ws_size;
  const float* x  = (const float*)d_in[0];
  const float* Wq = (const float*)d_in[1];
  const float* bq = (const float*)d_in[2];
  const float* Wk = (const float*)d_in[3];
  const float* bk = (const float*)d_in[4];
  const float* Wv = (const float*)d_in[5];
  const float* bv = (const float*)d_in[6];
  const float* Wo = (const float*)d_in[7];
  const float* bo = (const float*)d_in[8];
  float* out = (float*)d_out;

  // workspace carve-up (bf16 elements); total = 24M ushort = 48 MB
  unsigned short* ws = (unsigned short*)d_ws;
  size_t off = 0;
  unsigned short* xb  = ws + off; off += (size_t)MROWS * EMB;  // x bf16
  unsigned short* wqb = ws + off; off += (size_t)EMB * EMB;
  unsigned short* wkb = ws + off; off += (size_t)EMB * EMB;
  unsigned short* wvb = ws + off; off += (size_t)EMB * EMB;
  unsigned short* wob = ws + off; off += (size_t)EMB * EMB;
  unsigned short* qb  = ws + off; off += (size_t)MROWS * EMB;
  unsigned short* kb  = ws + off; off += (size_t)MROWS * EMB;
  unsigned short* vb  = ws + off; off += (size_t)MROWS * EMB;
  unsigned short* ob  = ws + off; off += (size_t)MROWS * EMB;

  // 1) bf16 casts (+ head-weight re-layout to [E, H*D])
  {
    const int nx = MROWS * EMB;                       // 4M
    k_conv_plain<<<(nx + 255) / 256, 256, 0, stream>>>(x, xb, nx);
    const int nw = EMB * EMB;                         // 1M
    k_conv_wqkv<<<(nw + 255) / 256, 256, 0, stream>>>(Wq, wqb);
    k_conv_wqkv<<<(nw + 255) / 256, 256, 0, stream>>>(Wk, wkb);
    k_conv_wqkv<<<(nw + 255) / 256, 256, 0, stream>>>(Wv, wvb);
    k_conv_plain<<<(nw + 255) / 256, 256, 0, stream>>>(Wo, wob, nw);
  }

  // 2) QKV projections: [4096,1024] @ [1024,1024] + bias -> bf16
  dim3 gg(MROWS / 128, EMB / 128);
  k_gemm<0><<<gg, 256, 0, stream>>>(xb, wqb, bq, qb, MROWS, EMB, EMB);
  k_gemm<0><<<gg, 256, 0, stream>>>(xb, wkb, bk, kb, MROWS, EMB, EMB);
  k_gemm<0><<<gg, 256, 0, stream>>>(xb, wvb, bv, vb, MROWS, EMB, EMB);

  // 3) fused causal attention (flash-style), output concat-heads bf16
  k_flash<<<BB * NH * (S_LEN / 128), 256, 0, stream>>>(qb, kb, vb, ob);

  // 4) output projection: [4096,1024] @ [1024,1024] + bo -> fp32 d_out
  k_gemm<1><<<gg, 256, 0, stream>>>(ob, wob, bo, out, MROWS, EMB, EMB);
}
